// MikesGraphNet_30408368456386
// MI455X (gfx1250) — matrix operand order repaired and checked by reading, not compiled, not run on hardware
//
#include <hip/hip_runtime.h>
#include <math.h>

typedef __attribute__((ext_vector_type(16))) __bf16 v16bf;
typedef __attribute__((ext_vector_type(8)))  float  v8f;

union Frag16 {
    v16bf v;
    uint4 q[2];
    unsigned short u[16];
};

__device__ __forceinline__ unsigned short f2bf(float x) {
    unsigned int u = __float_as_uint(x);
    unsigned int r = u + 0x7FFFu + ((u >> 16) & 1u);   // round-to-nearest-even
    return (unsigned short)(r >> 16);
}

__device__ __forceinline__ float gelu_tanh(float x) {
    float t = 0.7978845608028654f * (x + 0.044715f * x * x * x);
    return 0.5f * x * (1.0f + tanhf(t));
}

__device__ __forceinline__ v8f wmma_bf16(v16bf a, v16bf b, v8f c) {
    return __builtin_amdgcn_wmma_f32_16x16x32_bf16(
        false, a, false, b, (short)0, c, false, false);
}

// A fragment for V_WMMA 16x16x32 bf16: row-major ushort tile in LDS,
// lane (l&15) = row M, lane group g selects K sub-chunks:
//   g=0: K = kbase+[0..7], kbase+[16..23]; g=1: K = kbase+[8..15], kbase+[24..31]
__device__ __forceinline__ v16bf load_a_tile(const unsigned short* tile, int stride,
                                             int kbase, int lane) {
    int g = lane >> 4, row = lane & 15;
    const unsigned short* p = tile + row * stride + kbase + g * 8;
    Frag16 f;
    f.q[0] = *(const uint4*)(p);
    f.q[1] = *(const uint4*)(p + 16);
    return f.v;
}

// B fragment from pre-packed weight buffer: 16 contiguous bf16 per lane.
__device__ __forceinline__ v16bf load_b_pk(const unsigned short* pk, int nt, int NK,
                                           int kk, int lane) {
    const unsigned short* p = pk + ((((nt * NK) + kk) * 32 + lane) << 4);
    Frag16 f;
    f.q[0] = *(const uint4*)(p);
    f.q[1] = *(const uint4*)(p + 8);
    return f.v;
}

// ---------------------------------------------------------------------------
// Pack W [K x 256] (row-major, f32) into per-lane WMMA B fragments (bf16).
// ---------------------------------------------------------------------------
__global__ void pack_w_kernel(const float* __restrict__ W, int NK,
                              unsigned short* __restrict__ out) {
    int i = blockIdx.x * 256 + threadIdx.x;
    int per = NK * 512;
    int total = 16 * per;
    if (i >= total) return;
    int nt = i / per;
    int rem = i % per;
    int kk = rem >> 9;
    int rem2 = rem & 511;
    int lane = rem2 >> 4;
    int j = rem2 & 15;
    int g = lane >> 4;
    int n = nt * 16 + (lane & 15);
    int k = kk * 32 + (j >> 3) * 16 + g * 8 + (j & 7);
    out[i] = f2bf(W[k * 256 + n]);
}

// ---------------------------------------------------------------------------
// Embedding block: h = gelu(concat(emb[type], x_extra) @ w_emb + b_emb)
// ---------------------------------------------------------------------------
__global__ __launch_bounds__(256) void embed_kernel(
    const int* __restrict__ atom_types, const float* __restrict__ x_extra,
    const float* __restrict__ emb_table, const float* __restrict__ w_emb,
    const float* __restrict__ b_emb, float* __restrict__ hF,
    unsigned short* __restrict__ hB) {
    __shared__ float f[48];
    int node = blockIdx.x;
    int t = threadIdx.x;
    if (t < 48) {
        f[t] = (t < 32) ? emb_table[atom_types[node] * 32 + t]
                        : x_extra[node * 16 + (t - 32)];
    }
    __syncthreads();
    float acc = b_emb[t];
#pragma unroll
    for (int k = 0; k < 48; ++k) acc = fmaf(f[k], w_emb[k * 256 + t], acc);
    float g = gelu_tanh(acc);
    hF[(size_t)node * 256 + t] = g;
    hB[(size_t)node * 256 + t] = f2bf(g);
}

// ---------------------------------------------------------------------------
// Generic node GEMM: out = act(A_bf16[rows x 256] @ Wpk + bias) + residual
// 16 rows per block, 8 waves x 2 N-tiles, K = 256 (8 WMMA k-steps).
// ---------------------------------------------------------------------------
__global__ __launch_bounds__(256) void node_gemm_kernel(
    const unsigned short* __restrict__ Abf, const unsigned short* __restrict__ pkB,
    const float* __restrict__ bias, const float* __restrict__ residual,
    float* __restrict__ outF, unsigned short* __restrict__ outBf, int act) {
    __shared__ __align__(16) unsigned short sA[16 * 256];
    int tid = threadIdx.x;
    int row0 = blockIdx.x * 16;

    {   // stage A tile: 16 rows x 256 bf16, 32B per thread
        int row = tid >> 4, c = (tid & 15) * 16;
        const uint4* s = (const uint4*)(Abf + (size_t)(row0 + row) * 256 + c);
        uint4* d = (uint4*)(sA + row * 256 + c);
        d[0] = s[0];
        d[1] = s[1];
    }
    __syncthreads();

    int lane = tid & 31;
    int g = lane >> 4;
    int wave = tid >> 5;

#pragma unroll
    for (int half = 0; half < 2; ++half) {
        int nt = wave * 2 + half;
        v8f acc = {};
#pragma unroll
        for (int kk = 0; kk < 8; ++kk) {
            v16bf a = load_a_tile(sA, 256, kk * 32, lane);
            v16bf b = load_b_pk(pkB, nt, 8, kk, lane);
            acc = wmma_bf16(a, b, acc);
        }
        int col = nt * 16 + (lane & 15);
        float bv = bias ? bias[col] : 0.0f;
#pragma unroll
        for (int r = 0; r < 8; ++r) {
            float v = acc[r] + bv;
            if (act == 1) v = gelu_tanh(v);
            int node = row0 + g * 8 + r;
            size_t idx = (size_t)node * 256 + col;
            if (residual) v += residual[idx];
            if (outF) outF[idx] = v;
            if (outBf) outBf[idx] = f2bf(v);
        }
    }
}

// ---------------------------------------------------------------------------
// Fused edge pipeline, 64 edges per workgroup (amortizes weight L2 traffic 4x
// vs 16-edge tiles; each B fragment in registers feeds 4 M-tile WMMAs).
// Dynamic LDS: 4 buffers x 64x256 bf16 = 128 KB (RBF tile aliases bP0).
// ---------------------------------------------------------------------------
#define TE 64

__device__ __forceinline__ void layer_lrelu_64(
    const unsigned short* Ain, const unsigned short* pk,
    const float* __restrict__ bias, unsigned short* out, int wave, int lane) {
    int g = lane >> 4, col15 = lane & 15;
#pragma unroll
    for (int half = 0; half < 2; ++half) {
        int nt = wave * 2 + half;
        v8f acc[4] = {};
#pragma unroll
        for (int kk = 0; kk < 8; ++kk) {
            v16bf b = load_b_pk(pk, nt, 8, kk, lane);
#pragma unroll
            for (int mt = 0; mt < 4; ++mt)
                acc[mt] = wmma_bf16(load_a_tile(Ain + mt * 4096, 256, kk * 32, lane),
                                    b, acc[mt]);
        }
        int col = nt * 16 + col15;
        float bv = bias[col];
#pragma unroll
        for (int mt = 0; mt < 4; ++mt)
#pragma unroll
            for (int r = 0; r < 8; ++r) {
                float v = acc[mt][r] + bv;
                v = (v > 0.0f) ? v : 0.01f * v;
                out[(mt * 16 + g * 8 + r) * 256 + col] = f2bf(v);
            }
    }
}

__global__ __launch_bounds__(256) void edge_mlp_kernel(
    const int* __restrict__ edge_index, const float* __restrict__ dists,
    const float* __restrict__ freq, const unsigned short* __restrict__ m_bf,
    const unsigned short* __restrict__ pk_r2m, const float* __restrict__ b_r2m,
    const unsigned short* __restrict__ pkW1a, const unsigned short* __restrict__ pkW1b,
    const unsigned short* __restrict__ pkW1c, const float* __restrict__ b1,
    const unsigned short* __restrict__ pkW2, const float* __restrict__ b2,
    const unsigned short* __restrict__ pkW3, const float* __restrict__ b3,
    const unsigned short* __restrict__ pkW4, const float* __restrict__ b4,
    float* __restrict__ agg, int E) {
    extern __shared__ __align__(16) unsigned short smem[];
    unsigned short* bSrc = smem;                 // TE*256
    unsigned short* bDst = smem + 1 * TE * 256;  // TE*256
    unsigned short* bAtt = smem + 2 * TE * 256;  // TE*256
    unsigned short* bP0  = smem + 3 * TE * 256;  // TE*256
    unsigned short* bRbf = bP0;                  // alias: consumed before bP0 written
    __shared__ int sSrcIdx[TE];
    __shared__ int sDstIdx[TE];

    int tid = threadIdx.x;
    int lane = tid & 31;
    int g = lane >> 4;
    int col15 = lane & 15;
    int wave = tid >> 5;

    if (tid < TE) {
        int e = blockIdx.x * TE + tid;
        sSrcIdx[tid] = edge_index[e];
        sDstIdx[tid] = edge_index[E + e];
    }
    __syncthreads();

    {   // gather m[src], m[dst] rows (L2-resident) into LDS: 128B per thread each
        int row = tid >> 2, c = (tid & 3) * 64;
        const uint4* s0 = (const uint4*)(m_bf + (size_t)sSrcIdx[row] * 256 + c);
        uint4* d0 = (uint4*)(bSrc + row * 256 + c);
#pragma unroll
        for (int q = 0; q < 8; ++q) d0[q] = s0[q];
        const uint4* s1 = (const uint4*)(m_bf + (size_t)sDstIdx[row] * 256 + c);
        uint4* d1 = (uint4*)(bDst + row * 256 + c);
#pragma unroll
        for (int q = 0; q < 8; ++q) d1[q] = s1[q];
    }
    {   // Bessel RBF with Envelope(p=6): 8 radial values per thread
        int row = tid >> 2;
        int r0 = (tid & 3) * 8;
        int e = blockIdx.x * TE + row;
        float d = dists[e] * (1.0f / 5.0f);
        float d2 = d * d;
        float d5 = d2 * d2 * d;
        float env = 1.0f / d - 28.0f * d5 + 48.0f * d5 * d - 21.0f * d5 * d2;
#pragma unroll
        for (int j = 0; j < 8; ++j)
            bRbf[row * 32 + r0 + j] = f2bf(env * __sinf(freq[r0 + j] * d));
    }
    __syncthreads();

    // edge_attr = rbf @ w_r2m + b_r2m   (K=32 -> one WMMA per tile)
#pragma unroll
    for (int half = 0; half < 2; ++half) {
        int nt = wave * 2 + half;
        v16bf b = load_b_pk(pk_r2m, nt, 1, 0, lane);
        int col = nt * 16 + col15;
        float bv = b_r2m[col];
#pragma unroll
        for (int mt = 0; mt < 4; ++mt) {
            v8f acc = {};
            acc = wmma_bf16(load_a_tile(bRbf + mt * 16 * 32, 32, 0, lane), b, acc);
#pragma unroll
            for (int r = 0; r < 8; ++r)
                bAtt[(mt * 16 + g * 8 + r) * 256 + col] = f2bf(acc[r] + bv);
        }
    }
    __syncthreads();   // also fences bRbf reads before layer 1 writes bP0 (alias)

    // Layer 1: lrelu(m_src@W1a + m_dst@W1b + attr@W1c + b1) -> bP0
#pragma unroll
    for (int half = 0; half < 2; ++half) {
        int nt = wave * 2 + half;
        v8f acc[4] = {};
#pragma unroll
        for (int kk = 0; kk < 8; ++kk) {
            v16bf b = load_b_pk(pkW1a, nt, 8, kk, lane);
#pragma unroll
            for (int mt = 0; mt < 4; ++mt)
                acc[mt] = wmma_bf16(load_a_tile(bSrc + mt * 4096, 256, kk * 32, lane),
                                    b, acc[mt]);
        }
#pragma unroll
        for (int kk = 0; kk < 8; ++kk) {
            v16bf b = load_b_pk(pkW1b, nt, 8, kk, lane);
#pragma unroll
            for (int mt = 0; mt < 4; ++mt)
                acc[mt] = wmma_bf16(load_a_tile(bDst + mt * 4096, 256, kk * 32, lane),
                                    b, acc[mt]);
        }
#pragma unroll
        for (int kk = 0; kk < 8; ++kk) {
            v16bf b = load_b_pk(pkW1c, nt, 8, kk, lane);
#pragma unroll
            for (int mt = 0; mt < 4; ++mt)
                acc[mt] = wmma_bf16(load_a_tile(bAtt + mt * 4096, 256, kk * 32, lane),
                                    b, acc[mt]);
        }
        int col = nt * 16 + col15;
        float bv = b1[col];
#pragma unroll
        for (int mt = 0; mt < 4; ++mt)
#pragma unroll
            for (int r = 0; r < 8; ++r) {
                float v = acc[mt][r] + bv;
                v = (v > 0.0f) ? v : 0.01f * v;
                bP0[(mt * 16 + g * 8 + r) * 256 + col] = f2bf(v);
            }
    }
    __syncthreads();

    // Layer 2: bP0 -> bSrc (reuse; m_src tile no longer needed)
    layer_lrelu_64(bP0, pkW2, b2, bSrc, wave, lane);
    __syncthreads();

    // Layer 3: bSrc -> bP0
    layer_lrelu_64(bSrc, pkW3, b3, bP0, wave, lane);
    __syncthreads();

    // Layer 4 (no activation) + atomic scatter-add into agg[dst]
#pragma unroll
    for (int half = 0; half < 2; ++half) {
        int nt = wave * 2 + half;
        v8f acc[4] = {};
#pragma unroll
        for (int kk = 0; kk < 8; ++kk) {
            v16bf b = load_b_pk(pkW4, nt, 8, kk, lane);
#pragma unroll
            for (int mt = 0; mt < 4; ++mt)
                acc[mt] = wmma_bf16(load_a_tile(bP0 + mt * 4096, 256, kk * 32, lane),
                                    b, acc[mt]);
        }
        int col = nt * 16 + col15;
        float bv = b4[col];
#pragma unroll
        for (int mt = 0; mt < 4; ++mt)
#pragma unroll
            for (int r = 0; r < 8; ++r) {
                int row = mt * 16 + g * 8 + r;
                int node = sDstIdx[row];
                atomicAdd(&agg[(size_t)node * 256 + col], acc[mt][r] + bv);
            }
    }
}

__global__ void f32_to_bf16_kernel(const float* __restrict__ in,
                                   unsigned short* __restrict__ out, int n) {
    int i = blockIdx.x * 256 + threadIdx.x;
    if (i < n) out[i] = f2bf(in[i]);
}

// ---------------------------------------------------------------------------
extern "C" void kernel_launch(void* const* d_in, const int* in_sizes, int n_in,
                              void* d_out, int out_size, void* d_ws, size_t ws_size,
                              hipStream_t stream) {
    const int* atom_types = (const int*)d_in[0];
    const float* x_extra = (const float*)d_in[1];
    const int* edge_index = (const int*)d_in[2];
    const float* dists = (const float*)d_in[3];
    const float* freq = (const float*)d_in[4];
    const float* emb_table = (const float*)d_in[5];
    const float* w_emb = (const float*)d_in[6];
    const float* b_emb = (const float*)d_in[7];
    const float* w_n2m = (const float*)d_in[8];
    const float* b_n2m = (const float*)d_in[9];
    const float* w_r2m = (const float*)d_in[10];
    const float* b_r2m = (const float*)d_in[11];
    const float* w_m2n = (const float*)d_in[12];
    const float* mlp_w1 = (const float*)d_in[13];
    const float* mlp_b1 = (const float*)d_in[14];
    const float* mlp_w2 = (const float*)d_in[15];
    const float* mlp_b2 = (const float*)d_in[16];
    const float* mlp_w3 = (const float*)d_in[17];
    const float* mlp_b3 = (const float*)d_in[18];
    const float* mlp_w4 = (const float*)d_in[19];
    const float* mlp_b4 = (const float*)d_in[20];
    const float* fc_w1 = (const float*)d_in[21];
    const float* fc_b1 = (const float*)d_in[22];
    const float* fc_w2 = (const float*)d_in[23];
    const float* fc_b2 = (const float*)d_in[24];

    const int N = in_sizes[0];
    const int E = in_sizes[3];

    // ---- carve workspace ----
    char* p = (char*)d_ws;
    auto carve = [&](size_t bytes) {
        void* r = (void*)p;
        p += (bytes + 255) & ~(size_t)255;
        return r;
    };
    unsigned short* pk_n2m = (unsigned short*)carve(256 * 256 * 2);
    unsigned short* pk_r2m = (unsigned short*)carve(32 * 256 * 2);
    unsigned short* pk_w1a = (unsigned short*)carve(256 * 256 * 2);
    unsigned short* pk_w1b = (unsigned short*)carve(256 * 256 * 2);
    unsigned short* pk_w1c = (unsigned short*)carve(256 * 256 * 2);
    unsigned short* pk_w2  = (unsigned short*)carve(256 * 256 * 2);
    unsigned short* pk_w3  = (unsigned short*)carve(256 * 256 * 2);
    unsigned short* pk_w4  = (unsigned short*)carve(256 * 256 * 2);
    unsigned short* pk_m2n = (unsigned short*)carve(256 * 256 * 2);
    unsigned short* pk_fc1 = (unsigned short*)carve(256 * 256 * 2);
    unsigned short* pk_fc2 = (unsigned short*)carve(256 * 256 * 2);
    float* hF  = (float*)carve((size_t)N * 256 * 4);
    unsigned short* hB = (unsigned short*)carve((size_t)N * 256 * 2);
    unsigned short* mB = (unsigned short*)carve((size_t)N * 256 * 2);
    float* aggF = (float*)carve((size_t)N * 256 * 4);
    unsigned short* aggB = (unsigned short*)carve((size_t)N * 256 * 2);
    float* h2F = (float*)carve((size_t)N * 256 * 4);
    unsigned short* h2B = (unsigned short*)carve((size_t)N * 256 * 2);
    unsigned short* fc1B = (unsigned short*)carve((size_t)N * 256 * 2);
    (void)ws_size; (void)n_in; (void)out_size;

    // ---- pack all GEMM weights into WMMA B-fragment layout ----
    pack_w_kernel<<<32 * 8, 256, 0, stream>>>(w_n2m, 8, pk_n2m);
    pack_w_kernel<<<32 * 1, 256, 0, stream>>>(w_r2m, 1, pk_r2m);
    pack_w_kernel<<<32 * 8, 256, 0, stream>>>(mlp_w1, 8, pk_w1a);
    pack_w_kernel<<<32 * 8, 256, 0, stream>>>(mlp_w1 + 256 * 256, 8, pk_w1b);
    pack_w_kernel<<<32 * 8, 256, 0, stream>>>(mlp_w1 + 512 * 256, 8, pk_w1c);
    pack_w_kernel<<<32 * 8, 256, 0, stream>>>(mlp_w2, 8, pk_w2);
    pack_w_kernel<<<32 * 8, 256, 0, stream>>>(mlp_w3, 8, pk_w3);
    pack_w_kernel<<<32 * 8, 256, 0, stream>>>(mlp_w4, 8, pk_w4);
    pack_w_kernel<<<32 * 8, 256, 0, stream>>>(w_m2n, 8, pk_m2n);
    pack_w_kernel<<<32 * 8, 256, 0, stream>>>(fc_w1, 8, pk_fc1);
    pack_w_kernel<<<32 * 8, 256, 0, stream>>>(fc_w2, 8, pk_fc2);

    // ---- h = gelu(concat(emb, x_extra) @ w_emb + b_emb) ----
    embed_kernel<<<N, 256, 0, stream>>>(atom_types, x_extra, emb_table,
                                        w_emb, b_emb, hF, hB);

    // ---- m = h @ w_n2m + b_n2m (bf16 out) ----
    node_gemm_kernel<<<N / 16, 256, 0, stream>>>(hB, pk_n2m, b_n2m,
                                                 nullptr, nullptr, mB, 0);

    // ---- agg = 0; fused edge MLP with atomic scatter ----
    hipMemsetAsync(aggF, 0, (size_t)N * 256 * 4, stream);
    edge_mlp_kernel<<<E / TE, 256, 4 * TE * 256 * 2, stream>>>(
        edge_index, dists, freq, mB, pk_r2m, b_r2m,
        pk_w1a, pk_w1b, pk_w1c, mlp_b1, pk_w2, mlp_b2, pk_w3, mlp_b3,
        pk_w4, mlp_b4, aggF, E);

    // ---- h2 = h + agg @ w_m2n ----
    f32_to_bf16_kernel<<<(N * 256 + 255) / 256, 256, 0, stream>>>(aggF, aggB, N * 256);
    node_gemm_kernel<<<N / 16, 256, 0, stream>>>(aggB, pk_m2n, nullptr,
                                                 hF, h2F, h2B, 0);

    // ---- FC residual: out = h2 + gelu(h2@fc_w1+b1)@fc_w2 + b2 ----
    node_gemm_kernel<<<N / 16, 256, 0, stream>>>(h2B, pk_fc1, fc_b1,
                                                 nullptr, nullptr, fc1B, 1);
    node_gemm_kernel<<<N / 16, 256, 0, stream>>>(fc1B, pk_fc2, fc_b2,
                                                 h2F, (float*)d_out, nullptr, 0);
}
